// TransformerBlock_65343632441396
// MI455X (gfx1250) — matrix operand order repaired and checked by reading, not compiled, not run on hardware
//
#include <hip/hip_runtime.h>
#include <math.h>

#define DEV __device__ __forceinline__

// ---------------------------------------------------------------------------
// Types for CDNA5 WMMA (gfx1250, wave32)
// ---------------------------------------------------------------------------
typedef __bf16 bf16_t;
typedef bf16_t v16bf __attribute__((ext_vector_type(16)));
typedef float  v8f   __attribute__((ext_vector_type(8)));

union FragU { uint4 q[2]; v16bf v; };

static constexpr int D_MODEL = 1024;
static constexpr int N_HEADS = 16;
static constexpr int D_HEAD  = 64;
static constexpr int D_FF    = 4096;
static constexpr int BATCH   = 2;
static constexpr int SEQ     = 2048;
static constexpr int ROWS    = BATCH * SEQ;     // 4096
static constexpr int QKV_W   = 3 * D_MODEL;     // 3072

DEV unsigned short f2bf(float f) {
  unsigned u = __float_as_uint(f);
  u += 0x7FFFu + ((u >> 16) & 1u);              // round to nearest even
  return (unsigned short)(u >> 16);
}

DEV v8f zero8() {
  v8f z;
#pragma unroll
  for (int e = 0; e < 8; ++e) z[e] = 0.0f;
  return z;
}

DEV v8f wmma_bf16(v16bf a, v16bf b, v8f c) {
  return __builtin_amdgcn_wmma_f32_16x16x32_bf16(
      false, a, false, b, (short)0, c, false, false);
}

// A-fragment (16x32, 16-bit): lane L: row m=L&15, kh=L>>4,
//   elements cover K = kh*8+{0..7} and 16+kh*8+{0..7} -> two b128 loads.
DEV v16bf load_frag_a(const unsigned short* p, int stride, int lane) {
  const int m = lane & 15, kh = lane >> 4;
  FragU f;
  f.q[0] = *(const uint4*)(p + (size_t)m * stride + kh * 8);
  f.q[1] = *(const uint4*)(p + (size_t)m * stride + kh * 8 + 16);
  return f.v;
}

// B-fragment (32x16, 16-bit): lane L: col n=L&15, kh=L>>4,
//   elements cover K = kh*8+{0..7} and 16+kh*8+{0..7} -> two b128 loads.
DEV v16bf load_frag_b(const unsigned short* p, int stride, int lane) {
  const int n = lane & 15, kh = lane >> 4;
  FragU f;
  f.q[0] = *(const uint4*)(p + (size_t)n * stride + kh * 8);
  f.q[1] = *(const uint4*)(p + (size_t)n * stride + kh * 8 + 16);
  return f.v;
}

// ---------------------------------------------------------------------------
// CDNA5 async Global->LDS copies (ASYNCcnt tracked).  Note: INST_OFFSET is
// added to BOTH the LDS address and the global address (ISA 08 §4.4), so one
// base pair covers a 64B copy via offset:0/16/32/48.
// ---------------------------------------------------------------------------
DEV unsigned lds_off(const void* p) { return (unsigned)(size_t)p; }

DEV void async_ld16(unsigned loff, const void* g) {
  asm volatile("global_load_async_to_lds_b128 %0, %1, off"
               :: "v"(loff), "v"(g) : "memory");
}
DEV void async_ld64(unsigned loff, const void* g) {
  asm volatile(
      "global_load_async_to_lds_b128 %0, %1, off\n\t"
      "global_load_async_to_lds_b128 %0, %1, off offset:16\n\t"
      "global_load_async_to_lds_b128 %0, %1, off offset:32\n\t"
      "global_load_async_to_lds_b128 %0, %1, off offset:48"
      :: "v"(loff), "v"(g) : "memory");
}
DEV void wait_async0() {
  asm volatile("s_wait_asynccnt 0x0" ::: "memory");
}

// ---------------------------------------------------------------------------
// fp32 -> bf16 conversion (weights)
// ---------------------------------------------------------------------------
__global__ void xfm_cvt_bf16_kernel(const float* __restrict__ in,
                                    unsigned short* __restrict__ out, int n) {
  int i = (blockIdx.x * 256 + threadIdx.x) * 4;
  if (i + 3 < n) {
    float4 v = *(const float4*)(in + i);
    out[i + 0] = f2bf(v.x);
    out[i + 1] = f2bf(v.y);
    out[i + 2] = f2bf(v.z);
    out[i + 3] = f2bf(v.w);
  }
}

// ---------------------------------------------------------------------------
// RMSNorm over D=1024, writes bf16.  One 256-thread block per row.
// ---------------------------------------------------------------------------
__global__ __launch_bounds__(256)
void xfm_rmsnorm_kernel(const float* __restrict__ x, const float* __restrict__ w,
                        unsigned short* __restrict__ out) {
  const int row = blockIdx.x;
  const int t = threadIdx.x;
  const float* xr = x + (size_t)row * D_MODEL;
  float4 v = *(const float4*)(xr + t * 4);
  float ss = v.x * v.x + v.y * v.y + v.z * v.z + v.w * v.w;
#pragma unroll
  for (int o = 16; o >= 1; o >>= 1) ss += __shfl_xor(ss, o, 32);
  __shared__ float partial[8];
  const int lane = t & 31, wv = t >> 5;
  if (lane == 0) partial[wv] = ss;
  __syncthreads();
  float tot = 0.0f;
#pragma unroll
  for (int i = 0; i < 8; ++i) tot += partial[i];
  const float inv = rsqrtf(tot * (1.0f / (float)D_MODEL) + 1e-5f);
  float4 wv4 = *(const float4*)(w + t * 4);
  const size_t o0 = (size_t)row * D_MODEL + t * 4;
  out[o0 + 0] = f2bf(v.x * inv * wv4.x);
  out[o0 + 1] = f2bf(v.y * inv * wv4.y);
  out[o0 + 2] = f2bf(v.z * inv * wv4.z);
  out[o0 + 3] = f2bf(v.w * inv * wv4.w);
}

// ---------------------------------------------------------------------------
// Tiled WMMA GEMM: C[M,N] = A[M,K](bf16) * W[N,K]^T(bf16), f32 accumulate.
// Block tile 128x128, 8 waves (4 along M, 2 along N), wave tile 32x64.
// Double-buffered LDS fed by async Global->LDS copies (no VGPR staging).
// EPI: 0 = f32, 1 = f32 + residual, 2 = bf16, 3 = gelu(erf) -> bf16
// ---------------------------------------------------------------------------
template <int EPI>
__global__ __launch_bounds__(256)
void xfm_gemm_bf16_kernel(const unsigned short* __restrict__ A,
                          const unsigned short* __restrict__ W,
                          const float* __restrict__ RES,
                          void* __restrict__ OUTP,
                          int M, int N, int K) {
  __shared__ __align__(16) unsigned short As[2][128][32];
  __shared__ __align__(16) unsigned short Bs[2][128][32];
  const int t = threadIdx.x, lane = t & 31, wv = t >> 5;
  const int wm = wv & 3, wn = wv >> 2;
  const int bm = blockIdx.y * 128, bn = blockIdx.x * 128;

  v8f acc[2][4];
#pragma unroll
  for (int i = 0; i < 2; ++i)
#pragma unroll
    for (int j = 0; j < 4; ++j) acc[i][j] = zero8();

  // Each thread owns one 64B row-chunk of either the A tile or the B tile.
  const int lr = t & 127;
  const unsigned short* gsrc =
      (t < 128) ? (A + (size_t)(bm + lr) * K) : (W + (size_t)(bn + lr) * K);
  const unsigned loff0 = lds_off((t < 128) ? &As[0][lr][0] : &Bs[0][lr][0]);
  const unsigned loff1 = lds_off((t < 128) ? &As[1][lr][0] : &Bs[1][lr][0]);

  async_ld64(loff0, gsrc);          // prime buffer 0
  int p = 0;
  for (int k0 = 0; k0 < K; k0 += 32) {
    wait_async0();
    __syncthreads();                 // tile p fully in LDS for all waves
    if (k0 + 32 < K) {
      async_ld64(p ? loff0 : loff1, gsrc + k0 + 32);   // fill other buffer
      __builtin_prefetch(gsrc + k0 + 64, 0, 0);        // tile-after-next -> L2
    }

    v16bf af[2], bfr[4];
#pragma unroll
    for (int i = 0; i < 2; ++i)
      af[i] = load_frag_a(&As[p][wm * 32 + i * 16][0], 32, lane);
#pragma unroll
    for (int j = 0; j < 4; ++j)
      bfr[j] = load_frag_b(&Bs[p][wn * 64 + j * 16][0], 32, lane);
#pragma unroll
    for (int i = 0; i < 2; ++i)
#pragma unroll
      for (int j = 0; j < 4; ++j)
        acc[i][j] = wmma_bf16(af[i], bfr[j], acc[i][j]);
    p ^= 1;
  }

  const int l15 = lane & 15, hh = lane >> 4;
#pragma unroll
  for (int i = 0; i < 2; ++i) {
#pragma unroll
    for (int j = 0; j < 4; ++j) {
#pragma unroll
      for (int r = 0; r < 8; ++r) {
        const int row = bm + wm * 32 + i * 16 + r + hh * 8;
        const int col = bn + wn * 64 + j * 16 + l15;
        const size_t idx = (size_t)row * N + col;
        float v = acc[i][j][r];
        if (EPI == 0) {
          ((float*)OUTP)[idx] = v;
        } else if (EPI == 1) {
          ((float*)OUTP)[idx] = v + RES[idx];
        } else if (EPI == 2) {
          ((unsigned short*)OUTP)[idx] = f2bf(v);
        } else {
          float g = 0.5f * v * (1.0f + erff(v * 0.70710678118654752f));
          ((unsigned short*)OUTP)[idx] = f2bf(g);
        }
      }
    }
  }
}

// ---------------------------------------------------------------------------
// Flash attention, causal.  Grid: (SEQ/128, BATCH*N_HEADS), 256 threads.
// Each wave owns 16 query rows; K/V streamed in 32-key tiles through LDS.
// K tile arrives via async Global->LDS; V needs a transpose so it is staged
// through VGPRs.  qkv layout: [b*S+s][which*1024 + h*64 + d] (bf16).
// ---------------------------------------------------------------------------
__global__ __launch_bounds__(256)
void xfm_attn_kernel(const unsigned short* __restrict__ qkv,
                     unsigned short* __restrict__ ctx) {
  __shared__ __align__(16) unsigned short Ks[32][64];   // [kpos][d]
  __shared__ __align__(16) unsigned short Vt[64][32];   // [d][kpos]
  __shared__ __align__(16) unsigned short Ps[8][16][32];

  const int t = threadIdx.x, lane = t & 31, wv = t >> 5;
  const int qblk = blockIdx.x, bh = blockIdx.y;
  const int b = bh >> 4, h = bh & 15;
  const int rowstride = QKV_W;
  const unsigned short* baseq = qkv + (size_t)b * SEQ * rowstride + h * D_HEAD;
  const unsigned short* basek = baseq + D_MODEL;
  const unsigned short* basev = baseq + 2 * D_MODEL;

  const int qbase = qblk * 128 + wv * 16;
  const int qmaxw = qbase + 15;
  const int l15 = lane & 15, hh = lane >> 4;

  // Q fragments straight from global (A-layout): rows qbase..qbase+15, d 0..63
  v16bf qf[2];
#pragma unroll
  for (int db = 0; db < 2; ++db)
    qf[db] = load_frag_a(baseq + (size_t)qbase * rowstride + db * 32, rowstride, lane);

  v8f o[4];
#pragma unroll
  for (int j = 0; j < 4; ++j) o[j] = zero8();
  float rm[8], rl[8];
#pragma unroll
  for (int r = 0; r < 8; ++r) { rm[r] = -1e30f; rl[r] = 0.0f; }

  const int krow = t >> 3, ch = t & 7;
  const unsigned kloff = lds_off(&Ks[krow][ch * 8]);

  const int nkb = qblk * 4 + 4;   // causal: only key blocks up to block's last q
  for (int kb = 0; kb < nkb; ++kb) {
    const int kst = kb * 32;
    // K tile: async copy straight into LDS.  V tile: transpose through VGPRs.
    async_ld16(kloff, basek + (size_t)(kst + krow) * rowstride + ch * 8);
    {
      union { uint4 q; unsigned short u[8]; } vu;
      vu.q = *(const uint4*)(basev + (size_t)(kst + krow) * rowstride + ch * 8);
#pragma unroll
      for (int e = 0; e < 8; ++e) Vt[ch * 8 + e][krow] = vu.u[e];
    }
    wait_async0();
    __syncthreads();

    if (kst <= qmaxw) {          // wave-uniform: EXEC stays all-ones for WMMA
      v8f s0 = zero8(), s1 = zero8();
#pragma unroll
      for (int db = 0; db < 2; ++db) {
        v16bf k0f = load_frag_b(&Ks[0][db * 32], 64, lane);
        v16bf k1f = load_frag_b(&Ks[16][db * 32], 64, lane);
        s0 = wmma_bf16(qf[db], k0f, s0);
        s1 = wmma_bf16(qf[db], k1f, s1);
      }
      // Online softmax across the 32 keys of this tile
#pragma unroll
      for (int r = 0; r < 8; ++r) {
        const int qpos = qbase + r + hh * 8;
        const int kp0 = kst + l15, kp1 = kst + 16 + l15;
        float a0 = s0[r] * 0.125f;
        float a1 = s1[r] * 0.125f;
        const bool m0 = kp0 > qpos, m1 = kp1 > qpos;
        if (m0) a0 = -1e30f;
        if (m1) a1 = -1e30f;
        float mx = fmaxf(a0, a1);
#pragma unroll
        for (int ofs = 1; ofs < 16; ofs <<= 1) mx = fmaxf(mx, __shfl_xor(mx, ofs, 32));
        const float mnew = fmaxf(rm[r], mx);
        const float corr = __expf(rm[r] - mnew);
        const float p0 = m0 ? 0.0f : __expf(a0 - mnew);
        const float p1 = m1 ? 0.0f : __expf(a1 - mnew);
        float ps = p0 + p1;
#pragma unroll
        for (int ofs = 1; ofs < 16; ofs <<= 1) ps += __shfl_xor(ps, ofs, 32);
        rl[r] = rl[r] * corr + ps;
        rm[r] = mnew;
#pragma unroll
        for (int j = 0; j < 4; ++j) o[j][r] *= corr;
        const int mr = r + hh * 8;
        Ps[wv][mr][l15]      = f2bf(p0);
        Ps[wv][mr][16 + l15] = f2bf(p1);
      }
      // O += P(16x32) * V(32x64)
      v16bf pf = load_frag_a(&Ps[wv][0][0], 32, lane);
#pragma unroll
      for (int j = 0; j < 4; ++j) {
        v16bf vf = load_frag_b(&Vt[j * 16][0], 32, lane);
        o[j] = wmma_bf16(pf, vf, o[j]);
      }
    }
    __syncthreads();
  }

  // Normalize and write ctx in model layout (bf16)
#pragma unroll
  for (int r = 0; r < 8; ++r) {
    const float inv = rl[r] > 0.0f ? 1.0f / rl[r] : 0.0f;
    const int q = qbase + r + hh * 8;
#pragma unroll
    for (int j = 0; j < 4; ++j) {
      ctx[(size_t)(b * SEQ + q) * D_MODEL + h * D_HEAD + j * 16 + l15] =
          f2bf(o[j][r] * inv);
    }
  }
}

// ---------------------------------------------------------------------------
// Host-side launcher
// ---------------------------------------------------------------------------
extern "C" void kernel_launch(void* const* d_in, const int* in_sizes, int n_in,
                              void* d_out, int out_size, void* d_ws, size_t ws_size,
                              hipStream_t stream) {
  (void)in_sizes; (void)n_in; (void)out_size; (void)ws_size;
  const float* x     = (const float*)d_in[0];
  const float* w_qkv = (const float*)d_in[1];
  const float* w_o   = (const float*)d_in[2];
  const float* ln1_w = (const float*)d_in[3];
  const float* ln2_w = (const float*)d_in[4];
  const float* w1    = (const float*)d_in[5];
  const float* w2    = (const float*)d_in[6];
  float* out = (float*)d_out;

  char* ws = (char*)d_ws;
  size_t off = 0;
  auto alloc = [&](size_t bytes) -> void* {
    void* p = ws + off;
    off = (off + bytes + 255) & ~(size_t)255;
    return p;
  };
  unsigned short* wqkv_bf  = (unsigned short*)alloc((size_t)QKV_W * D_MODEL * 2);
  unsigned short* wo_bf    = (unsigned short*)alloc((size_t)D_MODEL * D_MODEL * 2);
  unsigned short* w1_bf    = (unsigned short*)alloc((size_t)D_FF * D_MODEL * 2);
  unsigned short* w2_bf    = (unsigned short*)alloc((size_t)D_MODEL * D_FF * 2);
  unsigned short* h1_bf    = (unsigned short*)alloc((size_t)ROWS * D_MODEL * 2);
  unsigned short* qkv_bf   = (unsigned short*)alloc((size_t)ROWS * QKV_W * 2);
  unsigned short* ctx_bf   = (unsigned short*)alloc((size_t)ROWS * D_MODEL * 2);
  float*          firstsum = (float*)alloc((size_t)ROWS * D_MODEL * 4);
  unsigned short* h2_bf    = (unsigned short*)alloc((size_t)ROWS * D_MODEL * 2);
  unsigned short* ff1_bf   = (unsigned short*)alloc((size_t)ROWS * D_FF * 2);

  auto cvt = [&](const float* in, unsigned short* o, int n) {
    xfm_cvt_bf16_kernel<<<(n / 4 + 255) / 256, 256, 0, stream>>>(in, o, n);
  };
  cvt(w_qkv, wqkv_bf, QKV_W * D_MODEL);
  cvt(w_o,   wo_bf,   D_MODEL * D_MODEL);
  cvt(w1,    w1_bf,   D_FF * D_MODEL);
  cvt(w2,    w2_bf,   D_MODEL * D_FF);

  // h1 = rmsnorm(x, ln1_w)
  xfm_rmsnorm_kernel<<<ROWS, 256, 0, stream>>>(x, ln1_w, h1_bf);
  // qkv = h1 @ w_qkv^T   (bf16 out)
  xfm_gemm_bf16_kernel<2><<<dim3(QKV_W / 128, ROWS / 128), 256, 0, stream>>>(
      h1_bf, wqkv_bf, nullptr, qkv_bf, ROWS, QKV_W, D_MODEL);
  // ctx = flash_attention(qkv)
  xfm_attn_kernel<<<dim3(SEQ / 128, BATCH * N_HEADS), 256, 0, stream>>>(qkv_bf, ctx_bf);
  // first_sum = ctx @ w_o^T + x   (f32 out)
  xfm_gemm_bf16_kernel<1><<<dim3(D_MODEL / 128, ROWS / 128), 256, 0, stream>>>(
      ctx_bf, wo_bf, x, firstsum, ROWS, D_MODEL, D_MODEL);
  // h2 = rmsnorm(first_sum, ln2_w)
  xfm_rmsnorm_kernel<<<ROWS, 256, 0, stream>>>(firstsum, ln2_w, h2_bf);
  // ff1 = gelu(h2 @ w1^T)   (bf16 out)
  xfm_gemm_bf16_kernel<3><<<dim3(D_FF / 128, ROWS / 128), 256, 0, stream>>>(
      h2_bf, w1_bf, nullptr, ff1_bf, ROWS, D_FF, D_MODEL);
  // out = ff1 @ w2^T + first_sum   (f32 out)
  xfm_gemm_bf16_kernel<1><<<dim3(D_MODEL / 128, ROWS / 128), 256, 0, stream>>>(
      ff1_bf, w2_bf, firstsum, out, ROWS, D_MODEL, D_FF);
}
